// SugarPairformer_22101901705611
// MI455X (gfx1250) — compile-verified, matrix operand order checked by reading, not run on hardware
//
#include <hip/hip_runtime.h>
#include <math.h>

typedef __bf16 bh;
typedef bh    bh8   __attribute__((ext_vector_type(8)));
typedef bh    bh16  __attribute__((ext_vector_type(16)));
typedef float f32x8 __attribute__((ext_vector_type(8)));

#define G_CH 6
#define L_CH 64
#define DZ   128
#define DS   384
#define MZ   (G_CH*L_CH*L_CH)   /* 24576 */
#define MS   (G_CH*L_CH)        /* 384   */
#define NSEQ 512
#define GLY  128
#define LDSTR 40                 /* LDS row stride (bf16): 80B, 16B-aligned  */

enum { ACT_NONE=0, ACT_SIG=1, ACT_SILU=2 };

__device__ __forceinline__ float sp_sig(float x){ return 1.0f/(1.0f+__expf(-x)); }
__device__ __forceinline__ int sp_swap_row(int r){
  // r = g*4096 + i*64 + j  ->  g*4096 + j*64 + i
  return (r & ~4095) | ((r & 63) << 6) | ((r >> 6) & 63);
}
// Lane fragment for 16-bit WMMA: two contiguous 8-element K-runs.
__device__ __forceinline__ bh16 sp_frag(const bh* rowp, int kb0){
  bh8 lo = *(const bh8*)(rowp + kb0);
  bh8 hi = *(const bh8*)(rowp + kb0 + 16);
  return __builtin_shufflevector(lo, hi, 0,1,2,3,4,5,6,7,8,9,10,11,12,13,14,15);
}

// ---------------------------------------------------------------------------
// Gather / scatter between padded [B,N,*] tensors and chain-packed buffers.
// ---------------------------------------------------------------------------
__global__ void sp_gather_s(const float* __restrict__ S, float* __restrict__ sb){
  int e = blockIdx.x*blockDim.x + threadIdx.x;
  if (e >= MS*DS) return;
  int c = e % DS, r = e / DS;           // r = g*64+i
  int g = r >> 6, i = r & 63;
  sb[e] = S[(size_t)(GLY + g*64 + i)*DS + c];
}
__global__ void sp_scatter_s(const float* __restrict__ sb, float* __restrict__ So){
  int e = blockIdx.x*blockDim.x + threadIdx.x;
  if (e >= MS*DS) return;
  int c = e % DS, r = e / DS;
  int g = r >> 6, i = r & 63;
  So[(size_t)(GLY + g*64 + i)*DS + c] = sb[e];
}
__global__ void sp_gather_z(const float* __restrict__ Z, float* __restrict__ zb){
  int e = blockIdx.x*blockDim.x + threadIdx.x;
  if (e >= MZ*DZ) return;
  int d = e & 127, j = (e>>7)&63, i = (e>>13)&63, g = e>>19;
  zb[e] = Z[(((size_t)(GLY + g*64 + i))*NSEQ + (GLY + g*64 + j))*DZ + d];
}
__global__ void sp_scatter_z(const float* __restrict__ zb, float* __restrict__ Zo){
  int e = blockIdx.x*blockDim.x + threadIdx.x;
  if (e >= MZ*DZ) return;
  int d = e & 127, j = (e>>7)&63, i = (e>>13)&63, g = e>>19;
  Zo[(((size_t)(GLY + g*64 + i))*NSEQ + (GLY + g*64 + j))*DZ + d] = zb[e];
}

// ---------------------------------------------------------------------------
// Row LayerNorm: out_bf16[r] = LN(X[src_row]) * gamma + beta
// ---------------------------------------------------------------------------
__global__ __launch_bounds__(128)
void sp_ln_rows(const float* __restrict__ X, const float* __restrict__ gmm,
                const float* __restrict__ bta, bh* __restrict__ Ob,
                int D, int swap_src){
  int r  = blockIdx.x;
  int sr = swap_src ? sp_swap_row(r) : r;
  const float* x = X + (size_t)sr * D;
  __shared__ float s1[128], s2[128];
  float a = 0.f, b = 0.f;
  for (int c = threadIdx.x; c < D; c += 128){ float v = x[c]; a += v; b += v*v; }
  s1[threadIdx.x] = a; s2[threadIdx.x] = b; __syncthreads();
  for (int s = 64; s > 0; s >>= 1){
    if (threadIdx.x < s){ s1[threadIdx.x]+=s1[threadIdx.x+s]; s2[threadIdx.x]+=s2[threadIdx.x+s]; }
    __syncthreads();
  }
  float mu = s1[0]/(float)D;
  float rs = rsqrtf(s2[0]/(float)D - mu*mu + 1e-5f);
  for (int c = threadIdx.x; c < D; c += 128)
    Ob[(size_t)r*D + c] = (bh)(((x[c]-mu)*rs)*gmm[c] + bta[c]);
}

// ---------------------------------------------------------------------------
// Generic WMMA GEMM: Y = epi(A_bf16[M,K] @ W_f32[K,NT]).  NT compile-time:
// W loads get immediate offsets, bounds checks fold away for NT%64==0, index
// math strength-reduces.  64x64 tile / 256 threads (8 waves, 16x32 each),
// double-buffered LDS, one barrier per K-step, fragments = 2x16B ds loads.
// ---------------------------------------------------------------------------
template<int NT>
__global__ __launch_bounds__(256)
void sp_gemm_t(const bh* __restrict__ A, const float* __restrict__ W,
               const float* __restrict__ bias,
               float* __restrict__ Cf, bh* __restrict__ Cb,
               const float* __restrict__ mulp, const float* __restrict__ addp,
               int M, int K, int act, int swap_store){
  constexpr bool NG = (NT & 63) != 0;     // need column guards?
  __shared__ bh As[2][64][LDSTR];         // [buf][m][k]
  __shared__ bh Ws[2][64][LDSTR];         // [buf][n][k]  (transposed W tile)
  int tid  = threadIdx.x;
  int m0   = blockIdx.y * 64;
  int n0   = blockIdx.x * 64;
  int lane = tid & 31, wave = tid >> 5;
  int wm = wave & 3, wn = wave >> 2;      // wave tile: rows wm*16, cols wn*32
  int l  = lane & 15, hi = lane >> 4;
  f32x8 acc0 = {}, acc1 = {};
  int arow = tid >> 2,  acol = (tid & 3) * 8;     // A: 8 contiguous bf16/thread
  int wn8  = tid & 63,  wk8  = (tid >> 6) * 8;    // W: 8 K-strided f32/thread
  bool wok = !NG || (n0 + wn8) < NT;
  const bh*    ap = A + (size_t)(m0 + arow)*K + acol;
  const float* wp = W + (size_t)wk8*NT + (wok ? (n0 + wn8) : 0);

  for (int k0 = 0; k0 < K; k0 += 32){
    int p = (k0 >> 5) & 1;
    *(bh8*)&As[p][arow][acol] = *(const bh8*)ap;
    bh8 wv;
    #pragma unroll
    for (int u = 0; u < 8; ++u){
      float f = wp[(size_t)u*NT];                 // immediate offsets
      wv[u] = (bh)(wok ? f : 0.0f);
    }
    *(bh8*)&Ws[p][wn8][wk8] = wv;
    ap += 32;
    wp += (size_t)32*NT;
    if (k0 + 32 < K){
      __builtin_prefetch(ap, 0, 1);
      __builtin_prefetch(wp, 0, 1);
    }
    __syncthreads();                              // single barrier per step
    bh16 af = sp_frag(&As[p][wm*16 + l][0],      8*hi);
    bh16 b0 = sp_frag(&Ws[p][wn*32 + l][0],      8*hi);
    bh16 b1 = sp_frag(&Ws[p][wn*32 + 16 + l][0], 8*hi);
    acc0 = __builtin_amdgcn_wmma_f32_16x16x32_bf16(false, af, false, b0, (short)0, acc0, false, false);
    acc1 = __builtin_amdgcn_wmma_f32_16x16x32_bf16(false, af, false, b1, (short)0, acc1, false, false);
  }

  #pragma unroll
  for (int r = 0; r < 8; ++r){
    int gr = m0 + wm*16 + r + 8*hi;       // C/D layout: VGPR r -> M=r (+8 hi-half)
    #pragma unroll
    for (int t = 0; t < 2; ++t){
      int gc = n0 + wn*32 + t*16 + l;
      if (NG && gc >= NT) continue;
      float y = t ? acc1[r] : acc0[r];
      if (bias) y += bias[gc];
      if (act == ACT_SIG)  y = sp_sig(y);
      else if (act == ACT_SILU) y = y * sp_sig(y);
      int sr = swap_store ? sp_swap_row(gr) : gr;
      size_t idx = (size_t)sr * NT + gc;
      if (mulp) y *= mulp[idx];
      if (addp) y += addp[idx];
      if (Cf) Cf[idx] = y;
      if (Cb) Cb[idx] = (bh)y;
    }
  }
}

// ---------------------------------------------------------------------------
// Triangle-mul pre-pass: pv = p * sigmoid(g), repacked to [G,128,64,64] bf16
// (a from channels [0,128), b from [128,256)).  swap=1 -> incoming variant.
// ---------------------------------------------------------------------------
__global__ void sp_gate_transpose(const float* __restrict__ P, const float* __restrict__ Gs,
                                  bh* __restrict__ At, bh* __restrict__ Bt, int swap){
  int e = blockIdx.x*blockDim.x + threadIdx.x;
  if (e >= G_CH*128*64*64) return;
  int c = e & 63, r = (e>>6)&63, d = (e>>12)&127, g = e>>19;
  int m = g*4096 + (swap ? c*64 + r : r*64 + c);
  size_t pi = (size_t)m * 256;
  At[e] = (bh)(P[pi + d]       * Gs[pi + d]);
  Bt[e] = (bh)(P[pi + 128 + d] * Gs[pi + 128 + d]);
}

// ---------------------------------------------------------------------------
// Batched triangle einsum: for batch (g,d): O = A(64x64) @ B(64x64)^T,
// scattered to o[g,i,j,d] f32.  768 batches, one per block, all-WMMA.
// ---------------------------------------------------------------------------
__global__ __launch_bounds__(256)
void sp_tri_einsum(const bh* __restrict__ At, const bh* __restrict__ Bt,
                   float* __restrict__ O){
  __shared__ bh Als[64][72];
  __shared__ bh Bls[64][72];
  int tid = threadIdx.x;
  int batch = blockIdx.x;                  // g*128 + d
  int g = batch >> 7, d = batch & 127;
  size_t base = (size_t)batch * 64 * 64;
  int r = tid >> 2, c0 = (tid & 3) * 16;
  *(bh8*)&Als[r][c0]   = *(const bh8*)&At[base + r*64 + c0];
  *(bh8*)&Als[r][c0+8] = *(const bh8*)&At[base + r*64 + c0 + 8];
  *(bh8*)&Bls[r][c0]   = *(const bh8*)&Bt[base + r*64 + c0];
  *(bh8*)&Bls[r][c0+8] = *(const bh8*)&Bt[base + r*64 + c0 + 8];
  __syncthreads();
  int lane = tid & 31, wave = tid >> 5;
  int tm = wave & 3, tn = wave >> 2;       // cols tn*16 and (tn+2)*16
  int l = lane & 15, hi = lane >> 4;
  f32x8 acc0 = {}, acc1 = {};
  #pragma unroll
  for (int k0 = 0; k0 < 64; k0 += 32){
    bh16 af = sp_frag(&Als[tm*16 + l][0],      k0 + 8*hi);
    bh16 b0 = sp_frag(&Bls[tn*16 + l][0],      k0 + 8*hi);   // B[k][n]=Bt[n][k]
    bh16 b1 = sp_frag(&Bls[(tn+2)*16 + l][0],  k0 + 8*hi);
    acc0 = __builtin_amdgcn_wmma_f32_16x16x32_bf16(false, af, false, b0, (short)0, acc0, false, false);
    acc1 = __builtin_amdgcn_wmma_f32_16x16x32_bf16(false, af, false, b1, (short)0, acc1, false, false);
  }
  #pragma unroll
  for (int rr = 0; rr < 8; ++rr){
    int gi = tm*16 + rr + 8*hi;
    O[(((size_t)g*64 + gi)*64 + (tn*16 + l))*128 + d]     = acc0[rr];
    O[(((size_t)g*64 + gi)*64 + ((tn+2)*16 + l))*128 + d] = acc1[rr];
  }
}

// ---------------------------------------------------------------------------
// Triangle attention core: one block per (g,i,h); thread j does online
// softmax over k (64) with head dim 16, then multiplies by gate -> bf16 out.
// ---------------------------------------------------------------------------
__global__ __launch_bounds__(64)
void sp_attn_tri(const bh* __restrict__ Q, const bh* __restrict__ Kb,
                 const bh* __restrict__ Vb, const float* __restrict__ B8,
                 const float* __restrict__ Gate, bh* __restrict__ O){
  int blk = blockIdx.x;                    // g*512 + i*8 + h
  int h = blk & 7, i = (blk >> 3) & 63, g = blk >> 9;
  int rowbase = g*4096 + i*64;
  __shared__ float kl[64][16], vl[64][16];
  int t = threadIdx.x;
  {
    bh8 k0 = *(const bh8*)&Kb[(size_t)(rowbase+t)*DZ + h*16];
    bh8 k1 = *(const bh8*)&Kb[(size_t)(rowbase+t)*DZ + h*16 + 8];
    bh8 v0 = *(const bh8*)&Vb[(size_t)(rowbase+t)*DZ + h*16];
    bh8 v1 = *(const bh8*)&Vb[(size_t)(rowbase+t)*DZ + h*16 + 8];
    #pragma unroll
    for (int u = 0; u < 8; ++u){
      kl[t][u] = (float)k0[u]; kl[t][8+u] = (float)k1[u];
      vl[t][u] = (float)v0[u]; vl[t][8+u] = (float)v1[u];
    }
  }
  __syncthreads();
  int j = t;
  float q[16];
  {
    bh8 q0 = *(const bh8*)&Q[(size_t)(rowbase+j)*DZ + h*16];
    bh8 q1 = *(const bh8*)&Q[(size_t)(rowbase+j)*DZ + h*16 + 8];
    #pragma unroll
    for (int u = 0; u < 8; ++u){ q[u] = (float)q0[u]; q[8+u] = (float)q1[u]; }
  }
  float mx = -1e30f, den = 0.f, acc[16];
  #pragma unroll
  for (int d2 = 0; d2 < 16; ++d2) acc[d2] = 0.f;
  for (int kk = 0; kk < 64; ++kk){
    float s = 0.f;
    #pragma unroll
    for (int d2 = 0; d2 < 16; ++d2) s += q[d2]*kl[kk][d2];
    s = s*0.25f + B8[(size_t)(g*4096 + j*64 + kk)*8 + h];
    if (s > mx){
      float corr = __expf(mx - s);
      den *= corr;
      #pragma unroll
      for (int d2 = 0; d2 < 16; ++d2) acc[d2] *= corr;
      mx = s;
    }
    float p = __expf(s - mx);
    den += p;
    #pragma unroll
    for (int d2 = 0; d2 < 16; ++d2) acc[d2] += p*vl[kk][d2];
  }
  float inv = 1.0f/den;
  #pragma unroll
  for (int d2 = 0; d2 < 16; ++d2){
    size_t idx = (size_t)(rowbase+j)*DZ + h*16 + d2;
    O[idx] = (bh)(acc[d2]*inv*Gate[idx]);
  }
}

// ---------------------------------------------------------------------------
// Sequence attention with pair bias: one block per (g,h), head dim 24.
// ---------------------------------------------------------------------------
__global__ __launch_bounds__(64)
void sp_attn_seq(const bh* __restrict__ Q, const bh* __restrict__ Kb,
                 const bh* __restrict__ Vb, const float* __restrict__ Z16,
                 const float* __restrict__ Gate, bh* __restrict__ O){
  int blk = blockIdx.x;                    // g*16 + h
  int h = blk & 15, g = blk >> 4;
  int rowbase = g*64;
  __shared__ float kl[64][24], vl[64][24];
  int t = threadIdx.x;
  #pragma unroll
  for (int ch = 0; ch < 3; ++ch){
    bh8 kv = *(const bh8*)&Kb[(size_t)(rowbase+t)*DS + h*24 + ch*8];
    bh8 vv = *(const bh8*)&Vb[(size_t)(rowbase+t)*DS + h*24 + ch*8];
    #pragma unroll
    for (int u = 0; u < 8; ++u){ kl[t][ch*8+u] = (float)kv[u]; vl[t][ch*8+u] = (float)vv[u]; }
  }
  __syncthreads();
  int qi = t;
  float q[24];
  #pragma unroll
  for (int ch = 0; ch < 3; ++ch){
    bh8 qv = *(const bh8*)&Q[(size_t)(rowbase+qi)*DS + h*24 + ch*8];
    #pragma unroll
    for (int u = 0; u < 8; ++u) q[ch*8+u] = (float)qv[u];
  }
  const float scale = 0.20412414523193154f;   // 1/sqrt(24)
  float mx = -1e30f, den = 0.f, acc[24];
  #pragma unroll
  for (int d2 = 0; d2 < 24; ++d2) acc[d2] = 0.f;
  for (int kk = 0; kk < 64; ++kk){
    float s = 0.f;
    #pragma unroll
    for (int d2 = 0; d2 < 24; ++d2) s += q[d2]*kl[kk][d2];
    s = s*scale + Z16[(size_t)(g*4096 + qi*64 + kk)*16 + h];
    if (s > mx){
      float corr = __expf(mx - s);
      den *= corr;
      #pragma unroll
      for (int d2 = 0; d2 < 24; ++d2) acc[d2] *= corr;
      mx = s;
    }
    float p = __expf(s - mx);
    den += p;
    #pragma unroll
    for (int d2 = 0; d2 < 24; ++d2) acc[d2] += p*vl[kk][d2];
  }
  float inv = 1.0f/den;
  #pragma unroll
  for (int d2 = 0; d2 < 24; ++d2){
    size_t idx = (size_t)(rowbase+qi)*DS + h*24 + d2;
    O[idx] = (bh)(acc[d2]*inv*Gate[idx]);
  }
}

// ---------------------------------------------------------------------------
// Host orchestration
// ---------------------------------------------------------------------------
extern "C" void kernel_launch(void* const* d_in, const int* in_sizes, int n_in,
                              void* d_out, int out_size, void* d_ws, size_t ws_size,
                              hipStream_t stream){
  (void)in_sizes; (void)n_in; (void)out_size; (void)ws_size;
  auto Pf = [&](int i)->const float* { return (const float*)d_in[i]; };

  // ---- workspace arena (f32 unless noted) ----
  char* w = (char*)d_ws; size_t off = 0;
  auto carve = [&](size_t bytes)->char*{ char* p = w + off; off = (off + bytes + 255) & ~(size_t)255; return p; };
  float* zbuf  = (float*)carve((size_t)MZ*DZ*4);       // packed pair activations
  float* sbuf  = (float*)carve((size_t)MS*DS*4);       // packed seq activations
  bh*    xbf   = (bh*)   carve((size_t)MZ*DZ*2);       // LN output (bf16)
  bh*    qbf   = (bh*)   carve((size_t)MZ*DZ*2);
  bh*    kbf   = (bh*)   carve((size_t)MZ*DZ*2);
  bh*    vbf   = (bh*)   carve((size_t)MZ*DZ*2);
  bh*    oattb = (bh*)   carve((size_t)MZ*DZ*2);
  float* Abig  = (float*)carve((size_t)MZ*512*4);      // P|G pair, or U1
  float* Pbuf  = Abig;
  float* Gbuf  = Abig + (size_t)MZ*256;
  float* U1    = Abig;
  float* Obuf  = (float*)carve((size_t)MZ*DZ*4);       // einsum output
  float* GateF = (float*)carve((size_t)MZ*DZ*4);       // sigmoid gates (f32)
  bh*    X2    = (bh*)   carve((size_t)MZ*512*2);      // transition hidden (bf16)
  float* zbh   = (float*)carve((size_t)MZ*16*4);       // pair-bias projections
  bh*    snbf  = (bh*)   carve((size_t)MS*DS*2);
  bh*    qsb   = (bh*)   carve((size_t)MS*DS*2);
  bh*    ksb   = (bh*)   carve((size_t)MS*DS*2);
  bh*    vsb   = (bh*)   carve((size_t)MS*DS*2);
  bh*    osb   = (bh*)   carve((size_t)MS*DS*2);
  float* GateS = (float*)carve((size_t)MS*DS*4);
  bh* a_t   = qbf;   // aliases: q/k/v unused during tri_mul
  bh* b_t   = kbf;
  bh* olnbf = vbf;

  auto gemm = [&](const bh* A, const float* W_, const float* bias,
                  float* Cf, bh* Cb, const float* mulp, const float* addp,
                  int M, int N, int K, int act, int swap){
    dim3 grid((N + 63)/64, M/64);
    switch (N){
      case 8:    sp_gemm_t<8>   <<<grid,256,0,stream>>>(A,W_,bias,Cf,Cb,mulp,addp,M,K,act,swap); break;
      case 16:   sp_gemm_t<16>  <<<grid,256,0,stream>>>(A,W_,bias,Cf,Cb,mulp,addp,M,K,act,swap); break;
      case 128:  sp_gemm_t<128> <<<grid,256,0,stream>>>(A,W_,bias,Cf,Cb,mulp,addp,M,K,act,swap); break;
      case 256:  sp_gemm_t<256> <<<grid,256,0,stream>>>(A,W_,bias,Cf,Cb,mulp,addp,M,K,act,swap); break;
      case 384:  sp_gemm_t<384> <<<grid,256,0,stream>>>(A,W_,bias,Cf,Cb,mulp,addp,M,K,act,swap); break;
      case 512:  sp_gemm_t<512> <<<grid,256,0,stream>>>(A,W_,bias,Cf,Cb,mulp,addp,M,K,act,swap); break;
      case 1536: sp_gemm_t<1536><<<grid,256,0,stream>>>(A,W_,bias,Cf,Cb,mulp,addp,M,K,act,swap); break;
      default: break;
    }
  };
  auto ln = [&](const float* X, const float* g_, const float* b_, bh* Ob, int M, int D, int swap){
    sp_ln_rows<<<M, 128, 0, stream>>>(X, g_, b_, Ob, D, swap);
  };

  // ---- gather chains ----
  sp_gather_s<<<(MS*DS + 255)/256, 256, 0, stream>>>(Pf(0), sbuf);
  sp_gather_z<<<(MZ*DZ + 255)/256, 256, 0, stream>>>(Pf(1), zbuf);

  for (int l = 0; l < 4; ++l){
    int pp = 5 + 57*l;
    int TMO = pp, TMI = pp+8, TAS = pp+16, TAE = pp+26, TRZ = pp+36, ATT = pp+41, TRS = pp+52;

    // --- triangle multiplications (outgoing, incoming) ---
    for (int v = 0; v < 2; ++v){
      int b = v ? TMI : TMO;  int swp = v;  // incoming -> swapped contraction
      ln(zbuf, Pf(b), Pf(b+1), xbf, MZ, DZ, 0);
      gemm(xbf, Pf(b+2), nullptr, Pbuf, nullptr, nullptr, nullptr, MZ, 256, DZ, ACT_NONE, 0); // w_p
      gemm(xbf, Pf(b+3), nullptr, Gbuf, nullptr, nullptr, nullptr, MZ, 256, DZ, ACT_SIG,  0); // w_g
      sp_gate_transpose<<<(G_CH*128*64*64)/256, 256, 0, stream>>>(Pbuf, Gbuf, a_t, b_t, swp);
      sp_tri_einsum<<<G_CH*128, 256, 0, stream>>>(a_t, b_t, Obuf);
      ln(Obuf, Pf(b+4), Pf(b+5), olnbf, MZ, DZ, 0);
      gemm(xbf, Pf(b+7), nullptr, GateF, nullptr, nullptr, nullptr, MZ, DZ, DZ, ACT_SIG, 0);  // w_go
      gemm(olnbf, Pf(b+6), nullptr, zbuf, nullptr, GateF, zbuf, MZ, DZ, DZ, ACT_NONE, 0);     // w_po, +res
    }

    // --- triangle attention (starting, ending) ---
    for (int v = 0; v < 2; ++v){
      int b = v ? TAE : TAS;  int swp = v;  // ending -> operate in transposed frame
      ln(zbuf, Pf(b), Pf(b+1), xbf, MZ, DZ, swp);
      gemm(xbf, Pf(b+2), nullptr, nullptr, qbf, nullptr, nullptr, MZ, DZ, DZ, ACT_NONE, 0);   // w_q
      gemm(xbf, Pf(b+3), nullptr, nullptr, kbf, nullptr, nullptr, MZ, DZ, DZ, ACT_NONE, 0);   // w_k
      gemm(xbf, Pf(b+4), nullptr, nullptr, vbf, nullptr, nullptr, MZ, DZ, DZ, ACT_NONE, 0);   // w_v
      gemm(xbf, Pf(b+5), nullptr, zbh,  nullptr, nullptr, nullptr, MZ, 8,  DZ, ACT_NONE, 0);  // w_b
      gemm(xbf, Pf(b+6), Pf(b+7), GateF, nullptr, nullptr, nullptr, MZ, DZ, DZ, ACT_SIG, 0);  // w_gate
      sp_attn_tri<<<G_CH*64*8, 64, 0, stream>>>(qbf, kbf, vbf, zbh, GateF, oattb);
      gemm(oattb, Pf(b+8), Pf(b+9), zbuf, nullptr, nullptr, zbuf, MZ, DZ, DZ, ACT_NONE, swp); // w_o, +res
    }

    // --- pair transition ---
    ln(zbuf, Pf(TRZ), Pf(TRZ+1), xbf, MZ, DZ, 0);
    gemm(xbf, Pf(TRZ+2), nullptr, U1, nullptr, nullptr, nullptr, MZ, 512, DZ, ACT_SILU, 0);   // w1
    gemm(xbf, Pf(TRZ+3), nullptr, nullptr, X2, U1, nullptr,      MZ, 512, DZ, ACT_NONE, 0);   // w2 * silu
    gemm(X2, Pf(TRZ+4), nullptr, zbuf, nullptr, nullptr, zbuf,   MZ, DZ, 512, ACT_NONE, 0);   // w3, +res

    // --- sequence attention with pair bias ---
    {
      int b = ATT;
      ln(sbuf, Pf(b), Pf(b+1), snbf, MS, DS, 0);
      gemm(snbf, Pf(b+2), Pf(b+3), nullptr, qsb, nullptr, nullptr, MS, DS, DS, ACT_NONE, 0);  // w_q + b_q
      gemm(snbf, Pf(b+4), nullptr, nullptr, ksb, nullptr, nullptr, MS, DS, DS, ACT_NONE, 0);  // w_k
      gemm(snbf, Pf(b+5), nullptr, nullptr, vsb, nullptr, nullptr, MS, DS, DS, ACT_NONE, 0);  // w_v
      gemm(snbf, Pf(b+6), nullptr, GateS, nullptr, nullptr, nullptr, MS, DS, DS, ACT_SIG, 0); // w_g
      ln(zbuf, Pf(b+7), Pf(b+8), xbf, MZ, DZ, 0);
      gemm(xbf, Pf(b+9), nullptr, zbh, nullptr, nullptr, nullptr, MZ, 16, DZ, ACT_NONE, 0);   // w_z
      sp_attn_seq<<<G_CH*16, 64, 0, stream>>>(qsb, ksb, vsb, zbh, GateS, osb);
      gemm(osb, Pf(b+10), nullptr, sbuf, nullptr, nullptr, sbuf, MS, DS, DS, ACT_NONE, 0);    // w_o, +res
    }

    // --- sequence transition ---
    ln(sbuf, Pf(TRS), Pf(TRS+1), xbf, MS, DS, 0);
    gemm(xbf, Pf(TRS+2), nullptr, U1, nullptr, nullptr, nullptr, MS, 1536, DS, ACT_SILU, 0);
    gemm(xbf, Pf(TRS+3), nullptr, nullptr, X2, U1, nullptr,      MS, 1536, DS, ACT_NONE, 0);
    gemm(X2, Pf(TRS+4), nullptr, sbuf, nullptr, nullptr, sbuf,   MS, DS, 1536, ACT_NONE, 0);
  }

  // ---- write outputs: copy inputs through, overwrite chain region ----
  float* outS = (float*)d_out;
  float* outZ = outS + (size_t)NSEQ*DS;
  hipMemcpyAsync(outS, d_in[0], (size_t)NSEQ*DS*4, hipMemcpyDeviceToDevice, stream);
  hipMemcpyAsync(outZ, d_in[1], (size_t)NSEQ*NSEQ*DZ*4, hipMemcpyDeviceToDevice, stream);
  sp_scatter_s<<<(MS*DS + 255)/256, 256, 0, stream>>>(sbuf, outS);
  sp_scatter_z<<<(MZ*DZ + 255)/256, 256, 0, stream>>>(zbuf, outZ);
}
